// NIMA_59579786330555
// MI455X (gfx1250) — matrix operand order repaired and checked by reading, not compiled, not run on hardware
//
#include <hip/hip_runtime.h>

typedef __attribute__((ext_vector_type(16))) _Float16 v16h;
typedef __attribute__((ext_vector_type(8)))  float    v8f;

#define HB    16
#define HWD   512
#define NPIX  (HWD*HWD)      // 262144
#define NB2   64
#define NB1   8
#define CIN   9
#define C1C   128
#define S1    64             // inner conv1 spatial (64x64)
#define POOL  33
#define C2C   1024
#define KGEMM 1152           // 9*128
#define PIXB  (POOL*POOL)    // 1089
#define MROWS (HB*PIXB)      // 17424
#define MTILE 256            // block M tile (8 waves x 32 rows)

// ---------------- Kernel 1: per-batch histograms -> hist_inp [B,9,64,64] ----
__global__ __launch_bounds__(256) void k_hist(const float* __restrict__ x1,
                                              float* __restrict__ hist) {
  __shared__ float h2[NB2 * NB2];
  __shared__ float h1[NB1];
  __shared__ float tot2, tot1;
  const int b = blockIdx.x, tid = threadIdx.x;
  for (int i = tid; i < NB2 * NB2; i += blockDim.x) h2[i] = 0.f;
  if (tid < NB1) h1[tid] = 0.f;
  __syncthreads();
  const float* L  = x1 + (size_t)(b * 3 + 0) * NPIX;
  const float* A  = x1 + (size_t)(b * 3 + 1) * NPIX;
  const float* Bc = x1 + (size_t)(b * 3 + 2) * NPIX;
  for (int p = tid; p < NPIX; p += blockDim.x) {
    float a = A[p], bb = Bc[p], l = L[p];
    float va = (a + 1.f) * 0.5f, vb = (bb + 1.f) * 0.5f;
    if (a != 0.f && bb != 0.f && va >= 0.f && va <= 1.f && vb >= 0.f && vb <= 1.f) {
      int ia = min(NB2 - 1, max(0, (int)floorf(va * NB2)));
      int ib = min(NB2 - 1, max(0, (int)floorf(vb * NB2)));
      atomicAdd(&h2[ia * NB2 + ib], 1.f);
    }
    float vl = (l + 1.f) * 0.5f;
    if (l != 0.f && vl >= 0.f && vl <= 1.f) {
      int il = min(NB1 - 1, max(0, (int)floorf(vl * NB1)));
      atomicAdd(&h1[il], 1.f);
    }
  }
  __syncthreads();
  if (tid == 0) {
    float t1 = 0.f;
    for (int i = 0; i < NB1; ++i) t1 += h1[i];
    tot1 = t1;
    tot2 = 0.f;
  }
  float s = 0.f;
  for (int i = tid; i < NB2 * NB2; i += blockDim.x) s += h2[i];
  __syncthreads();
  atomicAdd(&tot2, s);
  __syncthreads();
  const float r2 = 1.f / tot2, r1 = 1.f / tot1;
  float* ob = hist + (size_t)b * CIN * NB2 * NB2;
  for (int i = tid; i < CIN * NB2 * NB2; i += blockDim.x) {
    int c = i / (NB2 * NB2), r = i % (NB2 * NB2);
    float v;
    if (c == 0) { int p = r / NB2, q = r % NB2; v = h2[q * NB2 + p] * r2; }  // transpose
    else        { v = h1[c - 1] * r1; }
    ob[i] = v;
  }
}

// ---------------- Kernel 2: conv1 (1x1, 9->128) + BN, inner 64x64 ----------
__global__ __launch_bounds__(256) void k_conv1(const float* __restrict__ hist,
                                               const float* __restrict__ w1,
                                               const float* __restrict__ g1, const float* __restrict__ b1,
                                               const float* __restrict__ m1, const float* __restrict__ v1,
                                               float* __restrict__ C1) {
  int idx = blockIdx.x * blockDim.x + threadIdx.x;
  if (idx >= HB * C1C * S1 * S1) return;
  int ww = idx & 63;
  int t  = idx >> 6;
  int hh = t & 63; t >>= 6;
  int c  = t & 127;
  int b  = t >> 7;
  float acc = 0.f;
  const float* hb = hist + (size_t)b * CIN * S1 * S1 + hh * S1 + ww;
#pragma unroll
  for (int i = 0; i < CIN; ++i) acc += w1[c * CIN + i] * hb[(size_t)i * S1 * S1];
  float inv = g1[c] * rsqrtf(v1[c] + 1e-5f);
  float sh  = b1[c] - m1[c] * inv;
  C1[idx] = acc * inv + sh;
}

// ---------------- Kernel 3: maxpool(3,2,1) on 66x66 + ReLU -> P f16 NHWC ----
__global__ __launch_bounds__(256) void k_pool(const float* __restrict__ C1,
                                              const float* __restrict__ g1, const float* __restrict__ b1,
                                              const float* __restrict__ m1, const float* __restrict__ v1,
                                              _Float16* __restrict__ P) {
  int idx = blockIdx.x * blockDim.x + threadIdx.x;
  if (idx >= HB * POOL * POOL * C1C) return;
  int c  = idx & 127;
  int t  = idx >> 7;
  int ow = t % POOL; t /= POOL;
  int oh = t % POOL;
  int b  = t / POOL;
  float inv = g1[c] * rsqrtf(v1[c] + 1e-5f);
  float sh  = b1[c] - m1[c] * inv;   // BN of zero-padded conv border
  float best = -3.4e38f;
  for (int dy = -1; dy <= 1; ++dy) {
    int h66 = 2 * oh + dy;
    if (h66 < 0 || h66 >= 66) continue;
    for (int dx = -1; dx <= 1; ++dx) {
      int w66 = 2 * ow + dx;
      if (w66 < 0 || w66 >= 66) continue;
      float v;
      if (h66 >= 1 && h66 <= 64 && w66 >= 1 && w66 <= 64)
        v = C1[(((size_t)b * C1C + c) * S1 + (h66 - 1)) * S1 + (w66 - 1)];
      else
        v = sh;
      best = fmaxf(best, v);
    }
  }
  best = fmaxf(best, 0.f);
  P[idx] = (_Float16)best;   // layout [b][oh][ow][c]
}

// ---------------- Kernel 4: w2 OIHW f32 -> Bmat [K=1152][N=1024] f16 --------
__global__ __launch_bounds__(256) void k_wconv(const float* __restrict__ w2,
                                               _Float16* __restrict__ Bh) {
  int idx = blockIdx.x * blockDim.x + threadIdx.x;
  if (idx >= KGEMM * C2C) return;
  int n  = idx & 1023;
  int k  = idx >> 10;
  int ci = k & 127;
  int pos = k >> 7;
  int ky = pos / 3, kx = pos % 3;
  Bh[idx] = (_Float16)w2[(((size_t)n * C1C + ci) * 3 + ky) * 3 + kx];
}

// ---------------- Kernel 5: zero the (batch,co) sum accumulator -------------
__global__ void k_zero(float* __restrict__ S, int n) {
  int i = blockIdx.x * blockDim.x + threadIdx.x;
  if (i < n) S[i] = 0.f;
}

// ---------------- Kernel 6: implicit-GEMM conv2 via WMMA + fused BN/ReLU/sum
// M = 17424 (b,oh,ow), N = 1024 (co), K = 1152 ((ky,kx,ci), ci innermost)
// block: 256 thr = 8 waves; tile 256M x 64N; wave: 32M x 64N (2x4 wmma accs)
// per k-step per wave: 4 A ds_load_b128 + 8 B ds_load_b128 -> 8 WMMAs
__global__ __launch_bounds__(256) void k_gemm(const _Float16* __restrict__ P,
                                              const _Float16* __restrict__ Bh,
                                              const float* __restrict__ g2, const float* __restrict__ b2,
                                              const float* __restrict__ m2, const float* __restrict__ v2,
                                              float* __restrict__ S) {
  __shared__ _Float16 As[MTILE][32];   // [m][k]  16 KB
  __shared__ _Float16 Bs[64][32];      // transposed: [n][k]  4 KB
  const int tid  = threadIdx.x;
  const int lane = tid & 31;
  const int wave = tid >> 5;
  const int nBase = blockIdx.x * 64;
  const int mBase = blockIdx.y * MTILE;
  v8f acc[2][4];
#pragma unroll
  for (int a = 0; a < 2; ++a)
#pragma unroll
    for (int s = 0; s < 4; ++s) acc[a][s] = (v8f){};

  for (int kk = 0; kk < KGEMM / 32; ++kk) {
    const int k0  = kk * 32;
    const int pos = k0 >> 7;         // (ky,kx): 32-chunk never straddles (128%32==0)
    const int ci0 = k0 & 127;
    const int ky = pos / 3, kx = pos % 3;
    __syncthreads();                 // protect previous iteration's LDS reads

    // ---- stage A tile: 256 rows x 32 halves (4 x uint4 per thread) ----
#pragma unroll
    for (int j = 0; j < 4; ++j) {
      int t   = tid + j * 256;
      int row = t >> 2, seg = t & 3;
      int m   = mBase + row;
      union { uint4 v; _Float16 h[8]; } tmp;
      tmp.v = make_uint4(0u, 0u, 0u, 0u);
      if (m < MROWS) {
        int b  = m / PIXB, pix = m % PIXB;
        int oh = pix / POOL, ow = pix % POOL;
        int ih = oh + ky - 1, iw = ow + kx - 1;
        if (ih >= 0 && ih < POOL && iw >= 0 && iw < POOL) {
          const _Float16* src = P + (((size_t)b * POOL + ih) * POOL + iw) * C1C + ci0 + seg * 8;
          tmp.v = *(const uint4*)src;
        }
      }
#pragma unroll
      for (int e = 0; e < 8; ++e) As[row][seg * 8 + e] = tmp.h[e];
    }

    // ---- stage B tile: 32 k-rows x 64 n-cols, stored [n][k] ----
    {
      int row = tid >> 3, seg = tid & 7;
      const _Float16* src = Bh + (size_t)(k0 + row) * C2C + nBase + seg * 8;
      union { uint4 v; _Float16 h[8]; } tmp;
      tmp.v = *(const uint4*)src;
#pragma unroll
      for (int e = 0; e < 8; ++e) Bs[seg * 8 + e][row] = tmp.h[e];
    }
    if (k0 + 32 < KGEMM)  // hint next B tile into cache
      __builtin_prefetch(Bh + (size_t)(k0 + 32) * C2C + nBase, 0, 1);
    __syncthreads();

    // ---- gather fragments per ISA 16-bit A(16x32)/B(32x16) lane layouts ----
    const int ab = (lane < 16) ? 0 : 8;
    v16h af[2];
#pragma unroll
    for (int a = 0; a < 2; ++a) {
      const int mrow = wave * 32 + a * 16 + (lane & 15);
#pragma unroll
      for (int j = 0; j < 4; ++j) {
        af[a][2 * j]         = As[mrow][ab + 2 * j];
        af[a][2 * j + 1]     = As[mrow][ab + 2 * j + 1];
        af[a][8 + 2 * j]     = As[mrow][ab + 16 + 2 * j];
        af[a][8 + 2 * j + 1] = As[mrow][ab + 16 + 2 * j + 1];
      }
    }
    const int kb = (lane < 16) ? 0 : 16;
    const int nc = lane & 15;
#pragma unroll
    for (int s = 0; s < 4; ++s) {
      v16h bf;
#pragma unroll
      for (int e = 0; e < 16; ++e) bf[e] = Bs[s * 16 + nc][(e < 8) ? (ab + e) : (ab + 8 + e)];
      acc[0][s] = __builtin_amdgcn_wmma_f32_16x16x32_f16(false, af[0], false, bf, (short)0, acc[0][s], false, false);
      acc[1][s] = __builtin_amdgcn_wmma_f32_16x16x32_f16(false, af[1], false, bf, (short)0, acc[1][s], false, false);
    }
  }

  // ---- epilogue: BN + ReLU, partial per-batch sums, atomic into S[b,co] ----
#pragma unroll
  for (int s = 0; s < 4; ++s) {
    const int co = nBase + s * 16 + (lane & 15);
    const float inv = g2[co] * rsqrtf(v2[co] + 1e-5f);
    const float sh  = b2[co] - m2[co] * inv;
#pragma unroll
    for (int a = 0; a < 2; ++a) {
      int curb = -1;
      float sum = 0.f;
#pragma unroll
      for (int r = 0; r < 8; ++r) {
        int mm = mBase + wave * 32 + a * 16 + ((lane < 16) ? r : 8 + r);
        if (mm < MROWS) {
          float v = acc[a][s][r] * inv + sh;
          v = v > 0.f ? v : 0.f;
          int bb = mm / PIXB;
          if (bb != curb) {
            if (curb >= 0) atomicAdd(&S[(size_t)curb * C2C + co], sum);
            curb = bb; sum = 0.f;
          }
          sum += v;
        }
      }
      if (curb >= 0) atomicAdd(&S[(size_t)curb * C2C + co], sum);
    }
  }
}

// ---------------- Kernel 7: mean + linear (1024->256) + copy x --------------
__global__ __launch_bounds__(256) void k_final(const float* __restrict__ x,
                                               const float* __restrict__ S,
                                               const float* __restrict__ wl,
                                               const float* __restrict__ bl,
                                               float* __restrict__ out) {
  const int b = blockIdx.x, tid = threadIdx.x;
  for (int i = tid; i < 2048; i += blockDim.x)
    out[(size_t)b * 2304 + i] = x[(size_t)b * 2048 + i];
  for (int o = tid; o < 256; o += blockDim.x) {
    const float* sb = S + (size_t)b * C2C;
    const float* wo = wl + (size_t)o * C2C;
    float acc = 0.f;
    for (int k = 0; k < C2C; ++k) acc += sb[k] * wo[k];
    out[(size_t)b * 2304 + 2048 + o] = acc * (1.f / (float)PIXB) + bl[o];
  }
}

extern "C" void kernel_launch(void* const* d_in, const int* in_sizes, int n_in,
                              void* d_out, int out_size, void* d_ws, size_t ws_size,
                              hipStream_t stream) {
  const float* x  = (const float*)d_in[0];
  const float* x1 = (const float*)d_in[1];
  const float* w1 = (const float*)d_in[2];
  const float* g1 = (const float*)d_in[3];
  const float* b1 = (const float*)d_in[4];
  const float* m1 = (const float*)d_in[5];
  const float* v1 = (const float*)d_in[6];
  const float* w2 = (const float*)d_in[7];
  const float* g2 = (const float*)d_in[8];
  const float* b2 = (const float*)d_in[9];
  const float* m2 = (const float*)d_in[10];
  const float* v2 = (const float*)d_in[11];
  const float* wl = (const float*)d_in[12];
  const float* bl = (const float*)d_in[13];
  float* out = (float*)d_out;

  // workspace carve-up (16B aligned)
  char* ws = (char*)d_ws;
  size_t off = 0;
  float*    hist = (float*)(ws + off);    off += (size_t)HB * CIN * NB2 * NB2 * 4;        // 2.36 MB
  float*    C1   = (float*)(ws + off);    off += (size_t)HB * C1C * S1 * S1 * 4;          // 33.6 MB
  _Float16* P    = (_Float16*)(ws + off); off += (size_t)HB * POOL * POOL * C1C * 2;      // 4.46 MB
  _Float16* Bh   = (_Float16*)(ws + off); off += (size_t)KGEMM * C2C * 2;                 // 2.36 MB
  float*    S    = (float*)(ws + off);    off += (size_t)HB * C2C * 4;                    // 64 KB

  k_hist<<<HB, 256, 0, stream>>>(x1, hist);
  k_conv1<<<(HB * C1C * S1 * S1) / 256, 256, 0, stream>>>(hist, w1, g1, b1, m1, v1, C1);
  k_pool<<<(HB * POOL * POOL * C1C + 255) / 256, 256, 0, stream>>>(C1, g1, b1, m1, v1, P);
  k_wconv<<<(KGEMM * C2C) / 256, 256, 0, stream>>>(w2, Bh);
  k_zero<<<(HB * C2C + 255) / 256, 256, 0, stream>>>(S, HB * C2C);
  dim3 gg(C2C / 64, (MROWS + MTILE - 1) / MTILE);
  k_gemm<<<gg, 256, 0, stream>>>(P, Bh, g2, b2, m2, v2, S);
  k_final<<<HB, 256, 0, stream>>>(x, S, wl, bl, out);
}
